// LPT_raw_5454608466701
// MI455X (gfx1250) — compile-verified, matrix operand order checked by reading, not run on hardware
//
#include <hip/hip_runtime.h>
#include <hip/hip_bf16.h>
#include <cstddef>
#include <cstdint>

typedef __attribute__((ext_vector_type(16))) _Float16 v16h;
typedef __attribute__((ext_vector_type(8)))  float    v8f;
typedef int v4i_t __attribute__((vector_size(16)));   // matches builtin param type

#define CIN   128
#define COUT  128
#define KNN   16
#define SLOPE 0.01f
#define EPS   1e-5f
#define WELEM (COUT * CIN)   // 16384 halves per weight matrix

// ---------------------------------------------------------------------------
// WMMA helpers (CDNA5 v_wmma_f32_16x16x32_f16, wave32)
// ---------------------------------------------------------------------------
__device__ __forceinline__ v8f wmma_f16(v16h a, v16h b, v8f c) {
  // 8 args: (neg_a, A, neg_b, B, c_mod, C, reuse_a, reuse_b)
  return __builtin_amdgcn_wmma_f32_16x16x32_f16(false, a, false, b, (short)0, c,
                                                false, false);
}

// A fragment (16x32 f16) for one k-chunk from an f32 row in global memory.
//   h[0..7]  = K kb..kb+7,  h[8..15] = K kb+16..kb+23,  kb = kc*32 + (lane>>4)*8
__device__ __forceinline__ v16h load_a_row_f32(const float* __restrict__ row,
                                               int kc, int lane) {
  int kb = kc * 32 + ((lane >> 4) << 3);
  const float4* p0 = reinterpret_cast<const float4*>(row + kb);
  const float4* p1 = reinterpret_cast<const float4*>(row + kb + 16);
  float4 a0 = p0[0], a1 = p0[1], b0 = p1[0], b1 = p1[1];
  union { v16h v; _Float16 h[16]; } u;
  u.h[0]  = (_Float16)a0.x; u.h[1]  = (_Float16)a0.y;
  u.h[2]  = (_Float16)a0.z; u.h[3]  = (_Float16)a0.w;
  u.h[4]  = (_Float16)a1.x; u.h[5]  = (_Float16)a1.y;
  u.h[6]  = (_Float16)a1.z; u.h[7]  = (_Float16)a1.w;
  u.h[8]  = (_Float16)b0.x; u.h[9]  = (_Float16)b0.y;
  u.h[10] = (_Float16)b0.z; u.h[11] = (_Float16)b0.w;
  u.h[12] = (_Float16)b1.x; u.h[13] = (_Float16)b1.y;
  u.h[14] = (_Float16)b1.z; u.h[15] = (_Float16)b1.w;
  return u.v;
}

// A fragment from stored f16 rows, fused per-channel affine + LeakyReLU.
__device__ __forceinline__ v16h load_a_w_affine(const _Float16* __restrict__ row,
                                                int kc, int lane,
                                                const float* __restrict__ scl,
                                                const float* __restrict__ shf) {
  int kb = kc * 32 + ((lane >> 4) << 3);
  union { v16h v; _Float16 h[16]; uint4 u4[2]; } u;
  u.u4[0] = *reinterpret_cast<const uint4*>(row + kb);
  u.u4[1] = *reinterpret_cast<const uint4*>(row + kb + 16);
#pragma unroll
  for (int j = 0; j < 16; ++j) {
    int c = kb + j + (j >= 8 ? 8 : 0);
    float x = (float)u.h[j] * scl[c] + shf[c];
    x = x >= 0.f ? x : SLOPE * x;
    u.h[j] = (_Float16)x;
  }
  return u.v;
}

// B fragment (32x16 f16): B[k][n] = W[n][k]; W is f16 row-major [128][128] in LDS.
__device__ __forceinline__ v16h load_b_lds(const _Float16* __restrict__ W,
                                           int ntile, int kc, int lane) {
  int n  = ntile * 16 + (lane & 15);
  int kb = kc * 32 + ((lane >> 4) << 3);
  const _Float16* p = W + n * CIN + kb;
  union { v16h v; _Float16 h[16]; uint4 u4[2]; } u;
  u.u4[0] = *reinterpret_cast<const uint4*>(p);
  u.u4[1] = *reinterpret_cast<const uint4*>(p + 16);
  return u.v;
}

// ---------------------------------------------------------------------------
// Stage one 128x128 f16 weight matrix (32KB) from global into LDS.
// Preferred path: GLOBAL_LOAD_ASYNC_TO_LDS_B128 (ASYNCcnt-tracked DMA);
// fallback: plain 16B vector copies. Caller must __syncthreads() after.
// ---------------------------------------------------------------------------
__device__ __forceinline__ void stage_w_lds(const _Float16* __restrict__ g,
                                            _Float16* s, int tid) {
#if __has_builtin(__builtin_amdgcn_global_load_async_to_lds_b128)
#pragma unroll
  for (int e = 0; e < 8; ++e) {
    int off = (tid + e * 256) * 8;   // 8 halves = 16 bytes per op
    __builtin_amdgcn_global_load_async_to_lds_b128(
        (__attribute__((address_space(1))) v4i_t*)(uintptr_t)(g + off),
        (__attribute__((address_space(3))) v4i_t*)(uintptr_t)(uint32_t)(uintptr_t)(s + off),
        0, 0);
  }
#if __has_builtin(__builtin_amdgcn_s_wait_asynccnt)
  __builtin_amdgcn_s_wait_asynccnt(0);
#else
  asm volatile("s_wait_asynccnt 0x0" ::: "memory");
#endif
#else
  const uint4* gs = reinterpret_cast<const uint4*>(g);
  uint4* ss = reinterpret_cast<uint4*>(s);
  for (int i = tid; i < WELEM / 8; i += 256) ss[i] = gs[i];
#endif
}

// ---------------------------------------------------------------------------
// Kernel W: one-time f32 -> f16 conversion of all four weight matrices
// dst layout: [0]=Wq, [1]=Wk, [2]=Wl, [3]=Wv, each 16384 halves
// ---------------------------------------------------------------------------
__global__ void cvt_weights_kernel(const float* __restrict__ Wq,
                                   const float* __restrict__ Wk,
                                   const float* __restrict__ Wl,
                                   const float* __restrict__ Wv,
                                   _Float16* __restrict__ dst) {
  int i = blockIdx.x * blockDim.x + threadIdx.x;   // 0 .. 16383
  dst[i]             = (_Float16)Wq[i];
  dst[WELEM + i]     = (_Float16)Wk[i];
  dst[2 * WELEM + i] = (_Float16)Wl[i];
  dst[3 * WELEM + i] = (_Float16)Wv[i];
}

// ---------------------------------------------------------------------------
// Kernel 0: brute-force kNN (K=16) with register insertion sort
// ---------------------------------------------------------------------------
#define KNN_TILE 512
__global__ void knn_kernel(const float* __restrict__ xyz_i,
                           const float* __restrict__ xyz_last,
                           int* __restrict__ idxbuf, int N) {
  __shared__ float sx[KNN_TILE], sy[KNN_TILE], sz[KNN_TILE];
  int tid = threadIdx.x;
  int qi  = blockIdx.x * blockDim.x + tid;
  float qx = 0.f, qy = 0.f, qz = 0.f;
  if (qi < N) { qx = xyz_last[qi*3+0]; qy = xyz_last[qi*3+1]; qz = xyz_last[qi*3+2]; }
  float bd[KNN]; int bi[KNN];
#pragma unroll
  for (int t = 0; t < KNN; ++t) { bd[t] = 3.4e38f; bi[t] = 0; }
  for (int base = 0; base < N; base += KNN_TILE) {
    int lim = min(KNN_TILE, N - base);
    __syncthreads();
    for (int t = tid; t < lim; t += blockDim.x) {
      sx[t] = xyz_i[(base + t)*3 + 0];
      sy[t] = xyz_i[(base + t)*3 + 1];
      sz[t] = xyz_i[(base + t)*3 + 2];
    }
    // prefetch next tile's coordinates into cache (global_prefetch_b8)
    if (base + KNN_TILE + tid < N)
      __builtin_prefetch(&xyz_i[(base + KNN_TILE + tid) * 3], 0, 3);
    __syncthreads();
    if (qi < N) {
      for (int j = 0; j < lim; ++j) {
        float dx = qx - sx[j], dy = qy - sy[j], dz = qz - sz[j];
        float d  = dx*dx + dy*dy + dz*dz;
        if (d < bd[KNN-1]) {
          float dc = d; int ic = base + j;
#pragma unroll
          for (int t = 0; t < KNN; ++t) {
            if (dc < bd[t]) {
              float td = bd[t]; int ti = bi[t];
              bd[t] = dc; bi[t] = ic; dc = td; ic = ti;
            }
          }
        }
      }
    }
  }
  if (qi < N) {
#pragma unroll
    for (int t = 0; t < KNN; ++t) idxbuf[qi*KNN + t] = bi[t];
  }
}

// ---------------------------------------------------------------------------
// Kernel 1: q = fea_last @ Wq^T + bq
// ---------------------------------------------------------------------------
__global__ void gemm_q_kernel(const float* __restrict__ fea_last,
                              const _Float16* __restrict__ Wq16,
                              const float* __restrict__ bq,
                              float* __restrict__ qbuf) {
  __shared__ _Float16 sW[WELEM];
  __shared__ float    sb[COUT];
  int tid = threadIdx.x, wave = tid >> 5, lane = tid & 31;
  int rowbase = blockIdx.x * 128 + wave * 16;
  const float* arow = fea_last + (size_t)(rowbase + (lane & 15)) * CIN;
  __builtin_prefetch(arow, 0, 3);
  __builtin_prefetch(arow + 64, 0, 3);
  stage_w_lds(Wq16, sW, tid);
  if (tid < COUT) sb[tid] = bq[tid];
  __syncthreads();

  v8f acc[8] = {};
#pragma unroll
  for (int kc = 0; kc < 4; ++kc) {
    v16h a = load_a_row_f32(arow, kc, lane);
#pragma unroll
    for (int nt = 0; nt < 8; ++nt)
      acc[nt] = wmma_f16(a, load_b_lds(sW, nt, kc, lane), acc[nt]);
  }
  int hi = lane >> 4, cl = lane & 15;
#pragma unroll
  for (int nt = 0; nt < 8; ++nt) {
    int c = nt*16 + cl;
#pragma unroll
    for (int r = 0; r < 8; ++r) {
      int m = r + hi*8;
      qbuf[(size_t)(rowbase + m)*COUT + c] = acc[nt][r] + sb[c];
    }
  }
}

// ---------------------------------------------------------------------------
// Kernel 2 (phase1): w = q[n] - (fk@Wk^T + bk); store f16; BN1 block partials
// ---------------------------------------------------------------------------
__global__ void phase1_kernel(const float* __restrict__ fea_i,
                              const int* __restrict__ idxbuf,
                              const float* __restrict__ qbuf,
                              const _Float16* __restrict__ Wk16,
                              const float* __restrict__ bk,
                              _Float16* __restrict__ wbuf,
                              float* __restrict__ partial) {
  __shared__ _Float16 sW[WELEM];
  __shared__ float    sb[COUT];
  __shared__ float    sredS[8*COUT], sredQ[8*COUT];
  int tid = threadIdx.x, wave = tid >> 5, lane = tid & 31;
  int tileIdx = blockIdx.x * 8 + wave;      // == point index n
  int rowbase = tileIdx * 16;
  const float* arow = fea_i + (size_t)idxbuf[rowbase + (lane & 15)] * CIN;
  __builtin_prefetch(arow, 0, 3);
  __builtin_prefetch(arow + 64, 0, 3);
  stage_w_lds(Wk16, sW, tid);
  if (tid < COUT) sb[tid] = bk[tid];
  __syncthreads();

  v8f acc[8] = {};
#pragma unroll
  for (int kc = 0; kc < 4; ++kc) {
    v16h a = load_a_row_f32(arow, kc, lane);
#pragma unroll
    for (int nt = 0; nt < 8; ++nt)
      acc[nt] = wmma_f16(a, load_b_lds(sW, nt, kc, lane), acc[nt]);
  }
  int hi = lane >> 4, cl = lane & 15;
#pragma unroll
  for (int nt = 0; nt < 8; ++nt) {
    int c = nt*16 + cl;
    float qv = qbuf[(size_t)tileIdx*COUT + c];
    float bias = sb[c];
    float s = 0.f, s2 = 0.f;
#pragma unroll
    for (int r = 0; r < 8; ++r) {
      int m = r + hi*8;
      float wv = qv - (acc[nt][r] + bias);
      wbuf[(size_t)(rowbase + m)*COUT + c] = (_Float16)wv;
      s += wv; s2 += wv*wv;
    }
    s  += __shfl_xor(s, 16, 32);
    s2 += __shfl_xor(s2, 16, 32);
    if (hi == 0) { sredS[wave*COUT + c] = s; sredQ[wave*COUT + c] = s2; }
  }
  __syncthreads();
  if (tid < COUT) {
    float S = 0.f, Q = 0.f;
    for (int w = 0; w < 8; ++w) { S += sredS[w*COUT + tid]; Q += sredQ[w*COUT + tid]; }
    partial[(size_t)blockIdx.x*256 + tid]       = S;
    partial[(size_t)blockIdx.x*256 + 128 + tid] = Q;
  }
}

// ---------------------------------------------------------------------------
// Kernel 3: fold partial sums -> BN scale/shift (deterministic fixed order)
// ---------------------------------------------------------------------------
__global__ void bn_stats_kernel(const float* __restrict__ partial, int nblocks,
                                float invcount,
                                const float* __restrict__ gamma,
                                const float* __restrict__ beta,
                                float* __restrict__ scale,
                                float* __restrict__ shift) {
  int c = threadIdx.x;
  float S = 0.f, Q = 0.f;
  for (int b = 0; b < nblocks; ++b) {
    S += partial[(size_t)b*256 + c];
    Q += partial[(size_t)b*256 + 128 + c];
  }
  float mean = S * invcount;
  float var  = Q * invcount - mean*mean;
  float inv  = rsqrtf(var + EPS);
  float g    = gamma[c] * inv;
  scale[c] = g;
  shift[c] = beta[c] - mean * g;
}

// ---------------------------------------------------------------------------
// Kernel 4 (phase2): h = leaky(bn1(w)); w2 = h@Wl^T + bl (in-place); BN2 partials
// ---------------------------------------------------------------------------
__global__ void phase2_kernel(_Float16* __restrict__ wbuf,
                              const _Float16* __restrict__ Wl16,
                              const float* __restrict__ bl,
                              const float* __restrict__ scale1,
                              const float* __restrict__ shift1,
                              float* __restrict__ partial) {
  __shared__ _Float16 sW[WELEM];
  __shared__ float    sb[COUT], ssc[COUT], ssh[COUT];
  __shared__ float    sredS[8*COUT], sredQ[8*COUT];
  int tid = threadIdx.x, wave = tid >> 5, lane = tid & 31;
  int rowbase = (blockIdx.x * 8 + wave) * 16;
  const _Float16* arow = wbuf + (size_t)(rowbase + (lane & 15)) * COUT;
  __builtin_prefetch(arow, 0, 3);
  stage_w_lds(Wl16, sW, tid);
  if (tid < COUT) { sb[tid] = bl[tid]; ssc[tid] = scale1[tid]; ssh[tid] = shift1[tid]; }
  __syncthreads();

  v8f acc[8] = {};
#pragma unroll
  for (int kc = 0; kc < 4; ++kc) {
    v16h a = load_a_w_affine(arow, kc, lane, ssc, ssh);
#pragma unroll
    for (int nt = 0; nt < 8; ++nt)
      acc[nt] = wmma_f16(a, load_b_lds(sW, nt, kc, lane), acc[nt]);
  }
  int hi = lane >> 4, cl = lane & 15;
#pragma unroll
  for (int nt = 0; nt < 8; ++nt) {
    int c = nt*16 + cl;
    float bias = sb[c];
    float s = 0.f, s2 = 0.f;
#pragma unroll
    for (int r = 0; r < 8; ++r) {
      int m = r + hi*8;
      float wv = acc[nt][r] + bias;
      wbuf[(size_t)(rowbase + m)*COUT + c] = (_Float16)wv;   // w2 over w
      s += wv; s2 += wv*wv;
    }
    s  += __shfl_xor(s, 16, 32);
    s2 += __shfl_xor(s2, 16, 32);
    if (hi == 0) { sredS[wave*COUT + c] = s; sredQ[wave*COUT + c] = s2; }
  }
  __syncthreads();
  if (tid < COUT) {
    float S = 0.f, Q = 0.f;
    for (int w = 0; w < 8; ++w) { S += sredS[w*COUT + tid]; Q += sredQ[w*COUT + tid]; }
    partial[(size_t)blockIdx.x*256 + tid]       = S;
    partial[(size_t)blockIdx.x*256 + 128 + tid] = Q;
  }
}

// ---------------------------------------------------------------------------
// Kernel 5 (phase3): v = fk@Wv^T + bv; y = leaky(bn2(w2)); softmax over K=16
// per channel via shfl_xor(16); out = sum_k soft * v.
// ---------------------------------------------------------------------------
__global__ void phase3_kernel(const float* __restrict__ fea_i,
                              const int* __restrict__ idxbuf,
                              const _Float16* __restrict__ w2buf,
                              const _Float16* __restrict__ Wv16,
                              const float* __restrict__ bv,
                              const float* __restrict__ scale2,
                              const float* __restrict__ shift2,
                              float* __restrict__ out) {
  __shared__ _Float16 sW[WELEM];
  __shared__ float    sb[COUT], ssc[COUT], ssh[COUT];
  int tid = threadIdx.x, wave = tid >> 5, lane = tid & 31;
  int n = blockIdx.x * 8 + wave;
  int rowbase = n * 16;
  const float* arow = fea_i + (size_t)idxbuf[rowbase + (lane & 15)] * CIN;
  __builtin_prefetch(arow, 0, 3);
  __builtin_prefetch(arow + 64, 0, 3);
  stage_w_lds(Wv16, sW, tid);
  if (tid < COUT) { sb[tid] = bv[tid]; ssc[tid] = scale2[tid]; ssh[tid] = shift2[tid]; }
  __syncthreads();

  v8f acc[8] = {};
#pragma unroll
  for (int kc = 0; kc < 4; ++kc) {
    v16h a = load_a_row_f32(arow, kc, lane);
#pragma unroll
    for (int nt = 0; nt < 8; ++nt)
      acc[nt] = wmma_f16(a, load_b_lds(sW, nt, kc, lane), acc[nt]);
  }
  int hi = lane >> 4, cl = lane & 15;
#pragma unroll
  for (int nt = 0; nt < 8; ++nt) {
    int c = nt*16 + cl;
    float sc = ssc[c], sh = ssh[c], bias = sb[c];
    float y[8];
#pragma unroll
    for (int r = 0; r < 8; ++r) {
      int m = r + hi*8;
      float x = (float)w2buf[(size_t)(rowbase + m)*COUT + c] * sc + sh;
      y[r] = x >= 0.f ? x : SLOPE * x;
    }
    float mx = y[0];
#pragma unroll
    for (int r = 1; r < 8; ++r) mx = fmaxf(mx, y[r]);
    mx = fmaxf(mx, __shfl_xor(mx, 16, 32));     // max over all 16 k
    float se = 0.f;
#pragma unroll
    for (int r = 0; r < 8; ++r) { y[r] = __expf(y[r] - mx); se += y[r]; }
    se += __shfl_xor(se, 16, 32);               // sum over all 16 k
    float dot = 0.f;
#pragma unroll
    for (int r = 0; r < 8; ++r) dot += y[r] * (acc[nt][r] + bias);
    dot += __shfl_xor(dot, 16, 32);             // sum over all 16 k
    dot /= se;
    if (hi == 0) out[(size_t)n*COUT + c] = dot;
  }
}

// ---------------------------------------------------------------------------
// Host launcher
// ---------------------------------------------------------------------------
extern "C" void kernel_launch(void* const* d_in, const int* in_sizes, int n_in,
                              void* d_out, int out_size, void* d_ws, size_t ws_size,
                              hipStream_t stream) {
  const float* fea_i    = (const float*)d_in[0];
  const float* fea_last = (const float*)d_in[1];
  const float* xyz_i    = (const float*)d_in[2];
  const float* xyz_last = (const float*)d_in[3];
  // d_in[4] = batch (all zeros, single batch) — unused
  const float* Wq = (const float*)d_in[5];  const float* bq = (const float*)d_in[6];
  const float* Wk = (const float*)d_in[7];  const float* bk = (const float*)d_in[8];
  const float* Wv = (const float*)d_in[9];  const float* bv = (const float*)d_in[10];
  const float* g1 = (const float*)d_in[11]; const float* b1 = (const float*)d_in[12];
  const float* Wl = (const float*)d_in[13]; const float* bl = (const float*)d_in[14];
  const float* g2 = (const float*)d_in[15]; const float* b2 = (const float*)d_in[16];
  float* out = (float*)d_out;

  const int N  = in_sizes[0] / CIN;      // 8192
  const int NB = N / 8;                  // phase blocks (8 tiles of 16 rows each)

  // workspace carve-up (all naturally aligned)
  char* ws = (char*)d_ws;
  int*       idxbuf = (int*)ws;        ws += (size_t)N * KNN * sizeof(int);
  _Float16*  wf16   = (_Float16*)ws;   ws += (size_t)4 * WELEM * sizeof(_Float16);
  float*     qbuf   = (float*)ws;      ws += (size_t)N * COUT * sizeof(float);
  _Float16*  wbuf   = (_Float16*)ws;   ws += (size_t)N * KNN * COUT * sizeof(_Float16);
  float*     part1  = (float*)ws;      ws += (size_t)NB * 256 * sizeof(float);
  float*     part2  = (float*)ws;      ws += (size_t)NB * 256 * sizeof(float);
  float*     scale1 = (float*)ws;      ws += COUT * sizeof(float);
  float*     shift1 = (float*)ws;      ws += COUT * sizeof(float);
  float*     scale2 = (float*)ws;      ws += COUT * sizeof(float);
  float*     shift2 = (float*)ws;      ws += COUT * sizeof(float);

  const float invcount = 1.0f / (float)((size_t)N * KNN);

  cvt_weights_kernel<<<WELEM / 256, 256, 0, stream>>>(Wq, Wk, Wl, Wv, wf16);
  knn_kernel   <<<(N + 255) / 256, 256, 0, stream>>>(xyz_i, xyz_last, idxbuf, N);
  gemm_q_kernel<<<N / 128,         256, 0, stream>>>(fea_last, wf16, bq, qbuf);
  phase1_kernel<<<NB,              256, 0, stream>>>(fea_i, idxbuf, qbuf,
                                                     wf16 + WELEM, bk, wbuf, part1);
  bn_stats_kernel<<<1, 128, 0, stream>>>(part1, NB, invcount, g1, b1, scale1, shift1);
  phase2_kernel<<<NB,              256, 0, stream>>>(wbuf, wf16 + 2*WELEM, bl,
                                                     scale1, shift1, part2);
  bn_stats_kernel<<<1, 128, 0, stream>>>(part2, NB, invcount, g2, b2, scale2, shift2);
  phase3_kernel<<<NB,              256, 0, stream>>>(fea_i, idxbuf, wbuf,
                                                     wf16 + 3*WELEM, bv,
                                                     scale2, shift2, out);
}
